// MultiFeatureGNN_18743237280336
// MI455X (gfx1250) — matrix operand order repaired and checked by reading, not compiled, not run on hardware
//
#include <hip/hip_runtime.h>
#include <stdint.h>

// ---------------- types ----------------
typedef __attribute__((ext_vector_type(16))) __bf16 v16bf;
typedef __attribute__((ext_vector_type(8)))  __bf16 v8bf;
typedef __attribute__((ext_vector_type(8)))  float  v8f;

#define DI __device__ __forceinline__
#define USE_ASYNC_COPY 1

DI unsigned short f2bf(float f) {
  union { float f; unsigned u; } x; x.f = f;
  unsigned u = x.u;
  if ((u & 0x7fffffffu) > 0x7f800000u) return (unsigned short)((u >> 16) | 0x0040u); // NaN
  u += 0x7fffu + ((u >> 16) & 1u);   // round-to-nearest-even
  return (unsigned short)(u >> 16);
}
DI float bf2f(unsigned short s) {
  union { unsigned u; float f; } x; x.u = ((unsigned)s) << 16; return x.f;
}
// monotone float<->uint encoding for atomicMax on signed floats
DI unsigned fenc(float f) {
  unsigned u = __float_as_uint(f);
  return (u & 0x80000000u) ? ~u : (u | 0x80000000u);
}
DI float fdec(unsigned u) {
  unsigned b = (u & 0x80000000u) ? (u & 0x7fffffffu) : ~u;
  return __uint_as_float(b);
}

#if USE_ASYNC_COPY
// CDNA5 async global->LDS copy (16B per lane). lds_off is an LDS byte offset
// (low 32 bits of the flat shared address: flat LDS addr = aperture | offset).
DI void async_g2l_b128(unsigned lds_off, const void* gaddr) {
  asm volatile("global_load_async_to_lds_b128 %0, %1, off"
               :: "v"(lds_off), "v"(gaddr) : "memory");
}
DI void wait_async0() { asm volatile("s_wait_asynccnt 0" ::: "memory"); }
#endif

enum { EP_F32 = 1, EP_BF16 = 2, EP_RELU = 4, EP_SEG = 8 };

union AV { v16bf v; v8bf h[2]; };

DI void loadA(AV& a, const unsigned short* aRow, int kc, int lhi) {
  int kbase = kc * 32 + lhi * 8;
  a.h[0] = *(const v8bf*)(aRow + kbase);       // K = kbase..+7
  a.h[1] = *(const v8bf*)(aRow + kbase + 16);  // K = kbase+16..+23
}
DI void loadB8(AV b[8], const unsigned short* Bbase, int kc, int lane) {
#pragma unroll
  for (int nt = 0; nt < 8; ++nt) {
    const unsigned short* bp = Bbase + ((size_t)(kc * 8 + nt) * 64 + lane) * 8;
    b[nt].h[0] = *(const v8bf*)bp;             // plane0 (lane stride 16B: conflict-free)
    b[nt].h[1] = *(const v8bf*)(bp + 256);     // plane1 (+512B)
  }
}
DI void wmma8(v8f acc[8], const AV& a, const AV b[8]) {
#pragma unroll
  for (int nt = 0; nt < 8; ++nt)
    acc[nt] = __builtin_amdgcn_wmma_f32_16x16x32_bf16(
        false, a.v, false, b[nt].v, (short)0, acc[nt], false, false);
}

// ---------------- WMMA GEMM: C[M,128] = act(A[M,K] @ B[K,128] + bias) ----------------
// A: bf16 row-major. Bp: prepacked, bank-conflict-free two-plane layout:
//   Bp[(((kc*8+nt)*2+p)*32 + lane)*8 + jj] =
//       bf16(W[kc*32 + 8*(lane>>4) + p*16 + jj][nt*16 + (lane&15)])
// One wave per 16-row tile; 8 col-tiles of 16 -> 128 cols; f32 accumulate.
// For K<=128 the whole packed B (K*256 bytes <= 32KB) is staged in LDS by the
// block via CDNA5 async copies; all 8 waves then read it with ds_load_b128.
__global__ void __launch_bounds__(256) gemm_bf16_n128(
    const unsigned short* A, const unsigned short* __restrict__ Bp,
    const float* __restrict__ bias, int M, int K,
    float* outF, unsigned short* outB,
    const int* __restrict__ seg, float* segOut, int segLd, int segOff,
    int flags)
{
  __shared__ unsigned short Bs[16384];        // 32KB: packed B for K<=128
  const bool useLds = (K <= 128);
  if (useLds) {
    // cooperative stage: K*16 chunks of 16B; K%32==0 -> chunk count is a
    // multiple of 512, so every iteration runs with full EXEC.
    unsigned nchunks = (unsigned)K * 16;
#if USE_ASYNC_COPY
    unsigned ldsBase = (unsigned)(uintptr_t)&Bs[0];
    for (unsigned idx = threadIdx.x; idx < nchunks; idx += 256)
      async_g2l_b128(ldsBase + idx * 16, (const char*)Bp + (size_t)idx * 16);
    wait_async0();
#else
    for (unsigned idx = threadIdx.x; idx < nchunks; idx += 256)
      *(uint4*)((char*)Bs + (size_t)idx * 16) =
          *(const uint4*)((const char*)Bp + (size_t)idx * 16);
#endif
    __syncthreads();
  }

  int wave = threadIdx.x >> 5;
  int lane = threadIdx.x & 31;
  int tile = blockIdx.x * (blockDim.x >> 5) + wave;
  if (tile * 16 >= M) return;                 // uniform per wave: EXEC stays all-1s
  int rowBase = tile * 16;
  int lhi = lane >> 4;                        // 0/1
  int llo = lane & 15;

  v8f acc[8] = {};
  int nkc = K >> 5;
  const unsigned short* aRow = A + (size_t)(rowBase + llo) * K;

  if (useLds) {
    if (nkc == 4) {                            // K=128: the hot case, fully unrolled,
      AV a0, a1;                               // ping-pong A (no copies, no WAR NOPs)
      loadA(a0, aRow, 0, lhi);
      { AV b[8]; loadB8(b, Bs, 0, lane); loadA(a1, aRow, 1, lhi); wmma8(acc, a0, b); }
      { AV b[8]; loadB8(b, Bs, 1, lane); loadA(a0, aRow, 2, lhi); wmma8(acc, a1, b); }
      { AV b[8]; loadB8(b, Bs, 2, lane); loadA(a1, aRow, 3, lhi); wmma8(acc, a0, b); }
      { AV b[8]; loadB8(b, Bs, 3, lane); wmma8(acc, a1, b); }
    } else {                                   // K=32: single chunk
      AV a0, b[8];
      loadA(a0, aRow, 0, lhi);
      loadB8(b, Bs, 0, lane);
      wmma8(acc, a0, b);
    }
  } else {                                     // K>128 (head GEMM only): direct global
    AV a0, a1;
    loadA(a0, aRow, 0, lhi);
    for (int kc = 0; kc < nkc; ++kc) {
      AV b[8];
      loadB8(b, Bp, kc, lane);
      if (kc + 1 < nkc) loadA(a1, aRow, kc + 1, lhi);
      wmma8(acc, a0, b);
      a0 = a1;
    }
  }

  // epilogue: lane holds col = nt*16 + llo, rows rowBase + 8*lhi + r (r=0..7)
#pragma unroll
  for (int nt = 0; nt < 8; ++nt) {
    int col = nt * 16 + llo;
    float bb = bias ? bias[col] : 0.f;
#pragma unroll
    for (int r = 0; r < 8; ++r) {
      int row = rowBase + lhi * 8 + r;
      float v = acc[nt][r] + bb;
      if (flags & EP_RELU) v = fmaxf(v, 0.f);
      if (flags & EP_F32)  outF[(size_t)row * 128 + col] = v;
      if (flags & EP_BF16) outB[(size_t)row * 128 + col] = f2bf(v);
      if (flags & EP_SEG)  atomicAdd(&segOut[(size_t)seg[row] * segLd + segOff + col], v);
    }
  }
}

// ---------------- weight prepack into two-plane WMMA B layout ----------------
__global__ void prepack_b(const float* __restrict__ W, int realK, int Kpad,
                          unsigned short* __restrict__ Bp)
{
  int idx = blockIdx.x * blockDim.x + threadIdx.x;
  int total = Kpad * 128;                     // (Kpad/32)*8*2*32*8
  if (idx >= total) return;
  int jj   = idx & 7;
  int lane = (idx >> 3) & 31;
  int p    = (idx >> 8) & 1;
  int nt   = (idx >> 9) & 7;
  int kc   = idx >> 12;
  int k = kc * 32 + (lane >> 4) * 8 + p * 16 + jj;
  int n = nt * 16 + (lane & 15);
  float v = (k < realK) ? W[(size_t)k * 128 + n] : 0.f;
  Bp[idx] = f2bf(v);
}

// ---------------- small utility kernels ----------------
__global__ void fill_u32(unsigned* p, unsigned v, size_t n) {
  size_t i = (size_t)blockIdx.x * blockDim.x + threadIdx.x;
  if (i < n) p[i] = v;
}
__global__ void cvt_f32_bf16(const float* __restrict__ in, unsigned short* __restrict__ out, size_t n) {
  size_t i = (size_t)blockIdx.x * blockDim.x + threadIdx.x;
  if (i < n) out[i] = f2bf(in[i]);
}
// edge_attr [E,16] f32 -> [E,32] bf16 (zero-padded K for WMMA)
__global__ void pad_edge_bf16(const float* __restrict__ ea, unsigned short* __restrict__ out, int E) {
  size_t i = (size_t)blockIdx.x * blockDim.x + threadIdx.x;
  if (i >= (size_t)E * 32) return;
  int col = (int)(i & 31);
  size_t row = i >> 5;
  out[i] = (col < 16) ? f2bf(ea[row * 16 + col]) : (unsigned short)0;
}
__global__ void add_to_bf16(const float* __restrict__ a, const float* __restrict__ b,
                            unsigned short* __restrict__ out, size_t n) {
  size_t i = (size_t)blockIdx.x * blockDim.x + threadIdx.x;
  if (i < n) out[i] = f2bf(a[i] + b[i]);
}

// msg = relu(h[src] + e), scatter-add into agg[dst]; thread = (edge, 4 cols)
__global__ void edge_msg(const float* __restrict__ h, const unsigned short* __restrict__ e,
                         const int* __restrict__ src, const int* __restrict__ dst,
                         float* agg, int E)
{
  long idx = (long)blockIdx.x * blockDim.x + threadIdx.x;
  if (idx >= (long)E * 32) return;
  int eid = (int)(idx >> 5);
  int c = (int)(idx & 31) * 4;
  int s = src[eid], d = dst[eid];
  float4 hv = *(const float4*)(h + (size_t)s * 128 + c);
  const unsigned short* ep = e + (size_t)eid * 128 + c;
  float m0 = fmaxf(hv.x + bf2f(ep[0]), 0.f);
  float m1 = fmaxf(hv.y + bf2f(ep[1]), 0.f);
  float m2 = fmaxf(hv.z + bf2f(ep[2]), 0.f);
  float m3 = fmaxf(hv.w + bf2f(ep[3]), 0.f);
  float* ag = agg + (size_t)d * 128 + c;
  atomicAdd(ag + 0, m0); atomicAdd(ag + 1, m1);
  atomicAdd(ag + 2, m2); atomicAdd(ag + 3, m3);
}

// LayerNorm over 128 features; one wave32 per row; optional residual R; f32 + optional bf16 out
__global__ void layernorm_rows(const float* A, const float* R,
                               const float* __restrict__ g, const float* __restrict__ bb,
                               int M, float* outF, int outLd, unsigned short* outB)
{
  int wave = threadIdx.x >> 5, lane = threadIdx.x & 31;
  int row = blockIdx.x * (blockDim.x >> 5) + wave;
  if (row >= M) return;
  int c = lane * 4;
  float4 v = *(const float4*)(A + (size_t)row * 128 + c);
  if (R) {
    float4 r = *(const float4*)(R + (size_t)row * 128 + c);
    v.x += r.x; v.y += r.y; v.z += r.z; v.w += r.w;
  }
  float s = v.x + v.y + v.z + v.w;
#pragma unroll
  for (int o = 16; o; o >>= 1) s += __shfl_xor(s, o);
  float mean = s * (1.f / 128.f);
  float d0 = v.x - mean, d1 = v.y - mean, d2 = v.z - mean, d3 = v.w - mean;
  float q = d0 * d0 + d1 * d1 + d2 * d2 + d3 * d3;
#pragma unroll
  for (int o = 16; o; o >>= 1) q += __shfl_xor(q, o);
  float inv = rsqrtf(q * (1.f / 128.f) + 1e-5f);
  float o0 = d0 * inv * g[c + 0] + bb[c + 0];
  float o1 = d1 * inv * g[c + 1] + bb[c + 1];
  float o2 = d2 * inv * g[c + 2] + bb[c + 2];
  float o3 = d3 * inv * g[c + 3] + bb[c + 3];
  float* op = outF + (size_t)row * outLd + c;
  op[0] = o0; op[1] = o1; op[2] = o2; op[3] = o3;
  if (outB) {
    unsigned short* ob = outB + (size_t)row * 128 + c;
    ob[0] = f2bf(o0); ob[1] = f2bf(o1); ob[2] = f2bf(o2); ob[3] = f2bf(o3);
  }
}

// attention: per (node, head) score + segmented max
__global__ void attn_scores(const float* __restrict__ k, const float* __restrict__ q,
                            const int* __restrict__ batch, float* scores, unsigned* smax, int N)
{
  int idx = blockIdx.x * blockDim.x + threadIdx.x;
  if (idx >= N * 4) return;
  int n = idx >> 2, hh = idx & 3;
  const float* kp = k + (size_t)n * 128 + hh * 32;
  const float* qp = q + hh * 32;
  float s = 0.f;
#pragma unroll
  for (int d = 0; d < 32; ++d) s += kp[d] * qp[d];
  s *= 0.17677669529663687f;                  // 1/sqrt(32)
  scores[idx] = s;
  atomicMax(&smax[batch[n] * 4 + hh], fenc(s));
}
__global__ void attn_ex(const float* __restrict__ scores, const unsigned* __restrict__ smax,
                        const int* __restrict__ batch, float* ex, float* den, int N)
{
  int idx = blockIdx.x * blockDim.x + threadIdx.x;
  if (idx >= N * 4) return;
  int n = idx >> 2, hh = idx & 3;
  int g = batch[n];
  float e = expf(scores[idx] - fdec(smax[g * 4 + hh]));
  ex[idx] = e;
  atomicAdd(&den[g * 4 + hh], e);
}
__global__ void attn_pool(const float* __restrict__ v, const float* __restrict__ ex,
                          const float* __restrict__ den, const int* __restrict__ batch,
                          float* pooled, int N)
{
  long idx = (long)blockIdx.x * blockDim.x + threadIdx.x;
  if (idx >= (long)N * 128) return;
  int n = (int)(idx >> 7);
  int c = (int)(idx & 127);
  int hh = c >> 5;
  int g = batch[n];
  float aw = ex[n * 4 + hh] / den[g * 4 + hh];
  atomicAdd(&pooled[(size_t)g * 128 + c], aw * v[idx]);
}

// final L2 normalize; one wave per row of 128
__global__ void l2norm_rows(const float* __restrict__ in, float* __restrict__ out, int M)
{
  int wave = threadIdx.x >> 5, lane = threadIdx.x & 31;
  int row = blockIdx.x * (blockDim.x >> 5) + wave;
  if (row >= M) return;
  int c = lane * 4;
  float4 v = *(const float4*)(in + (size_t)row * 128 + c);
  float q = v.x * v.x + v.y * v.y + v.z * v.z + v.w * v.w;
#pragma unroll
  for (int o = 16; o; o >>= 1) q += __shfl_xor(q, o);
  float inv = 1.f / fmaxf(sqrtf(q), 1e-12f);
  float* op = out + (size_t)row * 128 + c;
  op[0] = v.x * inv; op[1] = v.y * inv; op[2] = v.z * inv; op[3] = v.w * inv;
}

// ---------------- host launcher ----------------
extern "C" void kernel_launch(void* const* d_in, const int* in_sizes, int n_in,
                              void* d_out, int out_size, void* d_ws, size_t ws_size,
                              hipStream_t stream)
{
  const int N = in_sizes[0] / 32;     // ND = 32
  const int E = in_sizes[1] / 16;     // ED = 16
  const int G = 256, L = 4;

  const float* x    = (const float*)d_in[0];
  const float* ea   = (const float*)d_in[1];
  const int*   eidx = (const int*)d_in[2];
  const int*   batch= (const int*)d_in[3];
  const float* ne_w1 = (const float*)d_in[4];  const float* ne_b1 = (const float*)d_in[5];
  const float* ne_w2 = (const float*)d_in[6];  const float* ne_b2 = (const float*)d_in[7];
  const float* ee_w1 = (const float*)d_in[8];  const float* ee_b1 = (const float*)d_in[9];
  const float* ee_w2 = (const float*)d_in[10]; const float* ee_b2 = (const float*)d_in[11];
  const float* conv_w1 = (const float*)d_in[12]; const float* conv_b1 = (const float*)d_in[13];
  const float* conv_w2 = (const float*)d_in[14]; const float* conv_b2 = (const float*)d_in[15];
  const float* norm_g = (const float*)d_in[16];  const float* norm_b = (const float*)d_in[17];
  const float* ms_w = (const float*)d_in[18];    const float* ms_b = (const float*)d_in[19];
  const float* q    = (const float*)d_in[20];
  const float* k_w  = (const float*)d_in[21]; const float* k_b = (const float*)d_in[22];
  const float* v_w  = (const float*)d_in[23]; const float* v_b = (const float*)d_in[24];
  const float* o_w  = (const float*)d_in[25]; const float* o_b = (const float*)d_in[26];
  const float* r_g  = (const float*)d_in[27]; const float* r_b = (const float*)d_in[28];
  const float* p_w1 = (const float*)d_in[29]; const float* p_b1 = (const float*)d_in[30];
  const float* p_w2 = (const float*)d_in[31]; const float* p_b2 = (const float*)d_in[32];
  const float* p_w3 = (const float*)d_in[33]; const float* p_b3 = (const float*)d_in[34];

  const int* srcI = eidx;
  const int* dstI = eidx + E;

  // --- bump allocator on d_ws ---
  char* wsb = (char*)d_ws;
  size_t off = 0;
  auto alloc = [&](size_t bytes) -> void* {
    void* p = wsb + off;
    off = (off + bytes + 255) & ~(size_t)255;
    return p;
  };
  // packed weights (bf16, Kpad*128 elems each)
  unsigned short* ne_w1p = (unsigned short*)alloc(32 * 128 * 2);
  unsigned short* ne_w2p = (unsigned short*)alloc(128 * 128 * 2);
  unsigned short* ee_w1p = (unsigned short*)alloc(32 * 128 * 2);
  unsigned short* ee_w2p = (unsigned short*)alloc(128 * 128 * 2);
  unsigned short* conv_w1p[4], *conv_w2p[4], *ms_wp[4];
  for (int i = 0; i < L; ++i) {
    conv_w1p[i] = (unsigned short*)alloc(128 * 128 * 2);
    conv_w2p[i] = (unsigned short*)alloc(128 * 128 * 2);
    ms_wp[i]    = (unsigned short*)alloc(128 * 128 * 2);
  }
  unsigned short* k_wp  = (unsigned short*)alloc(128 * 128 * 2);
  unsigned short* v_wp  = (unsigned short*)alloc(128 * 128 * 2);
  unsigned short* o_wp  = (unsigned short*)alloc(128 * 128 * 2);
  unsigned short* p_w1p = (unsigned short*)alloc(640 * 128 * 2);
  unsigned short* p_w2p = (unsigned short*)alloc(128 * 128 * 2);
  unsigned short* p_w3p = (unsigned short*)alloc(128 * 128 * 2);
  // activations
  unsigned short* x_bf  = (unsigned short*)alloc((size_t)N * 32 * 2);
  unsigned short* ea_bf = (unsigned short*)alloc((size_t)E * 32 * 2);
  unsigned short* e_bf  = (unsigned short*)alloc((size_t)E * 128 * 2);  // edge feats (in-place chain)
  float*          h     = (float*)alloc((size_t)N * 128 * 4);
  unsigned short* h_bf  = (unsigned short*)alloc((size_t)N * 128 * 2);
  float*          agg   = (float*)alloc((size_t)N * 128 * 4);           // also conv2 f32 out
  unsigned short* t_bf  = (unsigned short*)alloc((size_t)N * 128 * 2);  // conv bf16 chain (in-place)
  float*          kbuf  = (float*)alloc((size_t)N * 128 * 4);
  float*          vbuf  = (float*)alloc((size_t)N * 128 * 4);
  float*          scores= (float*)alloc((size_t)N * 4 * 4);
  float*          exbuf = (float*)alloc((size_t)N * 4 * 4);
  unsigned*       smax  = (unsigned*)alloc(G * 4 * 4);
  float*          den   = (float*)alloc(G * 4 * 4);
  float*          pooled= (float*)alloc(G * 128 * 4);
  unsigned short* pooled_bf = (unsigned short*)alloc(G * 128 * 2);
  float*          gtmp  = (float*)alloc(G * 128 * 4);
  float*          P     = (float*)alloc(G * 640 * 4);
  unsigned short* P_bf  = (unsigned short*)alloc(G * 640 * 2);
  unsigned short* hb    = (unsigned short*)alloc(G * 128 * 2);
  float*          pre   = (float*)alloc(G * 128 * 4);
  (void)ws_size; (void)n_in; (void)out_size;

  auto gemm = [&](const unsigned short* A, const unsigned short* Bp, const float* bias,
                  int M, int K, int flags, float* outF, unsigned short* outB,
                  const int* seg, float* segOut, int segLd, int segOff) {
    int tiles = (M + 15) >> 4;
    int blocks = (tiles + 7) >> 3;
    gemm_bf16_n128<<<blocks, 256, 0, stream>>>(A, Bp, bias, M, K, outF, outB,
                                               seg, segOut, segLd, segOff, flags);
  };
  auto prepack = [&](const float* W, int realK, int Kpad, unsigned short* dst) {
    int total = Kpad * 128;
    prepack_b<<<(total + 255) / 256, 256, 0, stream>>>(W, realK, Kpad, dst);
  };
  auto zero = [&](void* p, size_t words) {
    fill_u32<<<(unsigned)((words + 255) / 256), 256, 0, stream>>>((unsigned*)p, 0u, words);
  };

  // ---- prepack all weights ----
  prepack(ne_w1, 32, 32, ne_w1p);   prepack(ne_w2, 128, 128, ne_w2p);
  prepack(ee_w1, 16, 32, ee_w1p);   prepack(ee_w2, 128, 128, ee_w2p);
  for (int i = 0; i < L; ++i) {
    prepack(conv_w1 + (size_t)i * 128 * 128, 128, 128, conv_w1p[i]);
    prepack(conv_w2 + (size_t)i * 128 * 128, 128, 128, conv_w2p[i]);
    prepack(ms_w    + (size_t)i * 128 * 128, 128, 128, ms_wp[i]);
  }
  prepack(k_w, 128, 128, k_wp); prepack(v_w, 128, 128, v_wp); prepack(o_w, 128, 128, o_wp);
  prepack(p_w1, 640, 640, p_w1p); prepack(p_w2, 128, 128, p_w2p); prepack(p_w3, 128, 128, p_w3p);

  // ---- input conversions ----
  {
    size_t n = (size_t)N * 32;
    cvt_f32_bf16<<<(unsigned)((n + 255) / 256), 256, 0, stream>>>(x, x_bf, n);
    size_t m = (size_t)E * 32;
    pad_edge_bf16<<<(unsigned)((m + 255) / 256), 256, 0, stream>>>(ea, ea_bf, E);
  }

  // ---- encoders ----
  gemm(x_bf, ne_w1p, ne_b1, N, 32, EP_RELU | EP_BF16, nullptr, t_bf, nullptr, nullptr, 0, 0);
  gemm(t_bf, ne_w2p, ne_b2, N, 128, EP_F32 | EP_BF16, h, h_bf, nullptr, nullptr, 0, 0);
  gemm(ea_bf, ee_w1p, ee_b1, E, 32, EP_RELU | EP_BF16, nullptr, e_bf, nullptr, nullptr, 0, 0);
  gemm(e_bf, ee_w2p, ee_b2, E, 128, EP_BF16, nullptr, e_bf, nullptr, nullptr, 0, 0); // in-place

  zero(P, (size_t)G * 640);

  // ---- GINE layers ----
  for (int i = 0; i < L; ++i) {
    zero(agg, (size_t)N * 128);
    {
      long total = (long)E * 32;
      edge_msg<<<(unsigned)((total + 255) / 256), 256, 0, stream>>>(h, e_bf, srcI, dstI, agg, E);
    }
    {
      size_t n = (size_t)N * 128;
      add_to_bf16<<<(unsigned)((n + 255) / 256), 256, 0, stream>>>(agg, h, t_bf, n);
    }
    gemm(t_bf, conv_w1p[i], conv_b1 + i * 128, N, 128, EP_RELU | EP_BF16,
         nullptr, t_bf, nullptr, nullptr, 0, 0);                           // in-place
    gemm(t_bf, conv_w2p[i], conv_b2 + i * 128, N, 128, EP_RELU | EP_F32,
         agg, nullptr, nullptr, nullptr, 0, 0);                            // t = relu(mlp)
    {
      int blocks = (N + 7) / 8;
      layernorm_rows<<<blocks, 256, 0, stream>>>(agg, h, norm_g + i * 128, norm_b + i * 128,
                                                 N, h, 128, h_bf);         // h = LN(t + res)
    }
    // multiscale projection fused with segment_sum into P[:, 128*(1+i) .. ]
    gemm(h_bf, ms_wp[i], ms_b + i * 128, N, 128, EP_SEG,
         nullptr, nullptr, batch, P, 640, 128 * (1 + i));
  }

  // ---- attention readout ----
  gemm(h_bf, k_wp, k_b, N, 128, EP_F32, kbuf, nullptr, nullptr, nullptr, 0, 0);
  gemm(h_bf, v_wp, v_b, N, 128, EP_F32, vbuf, nullptr, nullptr, nullptr, 0, 0);
  zero(smax, G * 4); zero(den, G * 4); zero(pooled, (size_t)G * 128);
  attn_scores<<<(N * 4 + 255) / 256, 256, 0, stream>>>(kbuf, q, batch, scores, smax, N);
  attn_ex<<<(N * 4 + 255) / 256, 256, 0, stream>>>(scores, smax, batch, exbuf, den, N);
  {
    long total = (long)N * 128;
    attn_pool<<<(unsigned)((total + 255) / 256), 256, 0, stream>>>(vbuf, exbuf, den, batch, pooled, N);
  }
  {
    size_t n = (size_t)G * 128;
    cvt_f32_bf16<<<(unsigned)((n + 255) / 256), 256, 0, stream>>>(pooled, pooled_bf, n);
  }
  gemm(pooled_bf, o_wp, o_b, G, 128, EP_F32, gtmp, nullptr, nullptr, nullptr, 0, 0);
  layernorm_rows<<<(G + 7) / 8, 256, 0, stream>>>(gtmp, nullptr, r_g, r_b, G, P, 640, nullptr);

  // ---- projection head ----
  {
    size_t n = (size_t)G * 640;
    cvt_f32_bf16<<<(unsigned)((n + 255) / 256), 256, 0, stream>>>(P, P_bf, n);
  }
  gemm(P_bf, p_w1p, p_b1, G, 640, EP_RELU | EP_BF16, nullptr, hb, nullptr, nullptr, 0, 0);
  gemm(hb, p_w2p, p_b2, G, 128, EP_RELU | EP_BF16, nullptr, hb, nullptr, nullptr, 0, 0); // in-place
  gemm(hb, p_w3p, p_b3, G, 128, EP_F32, pre, nullptr, nullptr, nullptr, 0, 0);
  l2norm_rows<<<(G + 7) / 8, 256, 0, stream>>>(pre, (float*)d_out, G);
}
